// IntentClassifier_2379411882503
// MI455X (gfx1250) — matrix operand order repaired and checked by reading, not compiled, not run on hardware
//
#include <hip/hip_runtime.h>
#include <math.h>

typedef __attribute__((ext_vector_type(16))) __bf16 v16bf;
typedef __attribute__((ext_vector_type(8)))  __bf16 v8bf;
typedef __attribute__((ext_vector_type(8)))  float  v8f;

#define TM 64
#define TN 128
#define TKB 64      // K-step per LDS tile
#define PITCH 72    // 64 + 8 pad (bf16 elems); row = 144 B (16-aligned)

#if defined(__AMDGCN__) && __has_builtin(__builtin_amdgcn_global_load_async_to_lds_b128)
#define USE_ASYNC_LDS 1
typedef int asy4i __attribute__((vector_size(16)));                 // GCC-vector int4 (builtin pointee type)
typedef __attribute__((address_space(1))) asy4i* asy_gp;            // global src
typedef __attribute__((address_space(3))) asy4i* asy_lp;            // LDS dst
#else
#define USE_ASYNC_LDS 0
#endif

__device__ __forceinline__ void wait_async0() {
#if __has_builtin(__builtin_amdgcn_s_wait_asynccnt)
    __builtin_amdgcn_s_wait_asynccnt(0);
#elif defined(__AMDGCN__)
    asm volatile("s_wait_asynccnt 0x0" ::: "memory");
#endif
}

// ---------------------------------------------------------------------------
// Generic bf16 WMMA GEMM:  C[M,N] = A[M,K] * Bt[N,K]^T   (A, Bt row-major bf16)
// Double-buffered LDS; async global->LDS when available; grouped WMMA issue.
// ---------------------------------------------------------------------------
template<bool OUT_BF16, bool DOGELU, bool PERM, bool VST>
__global__ __launch_bounds__(256) void wmma_gemm(
    const __bf16* __restrict__ A, const __bf16* __restrict__ Bt, void* __restrict__ Cout,
    int M, int N, int K,
    long sA, long sB, long sC,
    const float* __restrict__ bias,                 // [N] or null
    const float* __restrict__ rowbias, int rowdiv,  // [*,N] indexed by m/rowdiv, or null
    const float* __restrict__ residual,             // f32 [M,N], or null
    float scale)
{
    __shared__ __bf16 As[2][TM * PITCH];
    __shared__ __bf16 Bs[2][TN * PITCH];
    const int tid  = threadIdx.x;
    const int lane = tid & 31;
    const int wv   = tid >> 5;     // 0..7
    const int wrow = wv & 3;       // 16-row strip
    const int wcol = wv >> 2;      // 64-col half
    const int hi   = lane >> 4;
    const int l16  = lane & 15;
    const int z     = blockIdx.z;
    const int tileM = blockIdx.x * TM;
    const int tileN = blockIdx.y * TN;
    A  += (long)z * sA;
    Bt += (long)z * sB;

    // cooperative-load geometry: chunk = 8 bf16; tile row = 8 chunks
    const int lrow = tid >> 3;          // 0..31
    const int lcol = (tid & 7) * 8;     // 0..56
    const __bf16* Aptr = A  + (long)(tileM + lrow) * K + lcol;
    const __bf16* Bptr = Bt + (long)(tileN + lrow) * K + lcol;

    v8f acc[4];
    #pragma unroll
    for (int f = 0; f < 4; ++f)
        #pragma unroll
        for (int j = 0; j < 8; ++j) acc[f][j] = 0.0f;

#if USE_ASYNC_LDS
    // async DMA: global -> LDS without VGPR round trip (tracked by ASYNCcnt)
    auto async_tile = [&](int k0, int buf) {
        #pragma unroll
        for (int i = 0; i < 2; ++i)
            __builtin_amdgcn_global_load_async_to_lds_b128(
                (asy_gp)(void*)(Aptr + (long)i * 32 * K + k0),
                (asy_lp)(void*)&As[buf][(lrow + 32 * i) * PITCH + lcol],
                0, 0);
        #pragma unroll
        for (int i = 0; i < 4; ++i)
            __builtin_amdgcn_global_load_async_to_lds_b128(
                (asy_gp)(void*)(Bptr + (long)i * 32 * K + k0),
                (asy_lp)(void*)&Bs[buf][(lrow + 32 * i) * PITCH + lcol],
                0, 0);
    };
    async_tile(0, 0);
    wait_async0();
    __syncthreads();
#else
    v8bf ar[2], br[4];
    auto fetch = [&](int k0) {
        #pragma unroll
        for (int i = 0; i < 2; ++i) ar[i] = *(const v8bf*)(Aptr + (long)i * 32 * K + k0);
        #pragma unroll
        for (int i = 0; i < 4; ++i) br[i] = *(const v8bf*)(Bptr + (long)i * 32 * K + k0);
    };
    auto stash = [&](int buf) {
        #pragma unroll
        for (int i = 0; i < 2; ++i)
            *(v8bf*)&As[buf][(lrow + 32 * i) * PITCH + lcol] = ar[i];
        #pragma unroll
        for (int i = 0; i < 4; ++i)
            *(v8bf*)&Bs[buf][(lrow + 32 * i) * PITCH + lcol] = br[i];
    };
    fetch(0);
    stash(0);
    __syncthreads();
#endif

    int cur = 0;
    const int ra = wrow * 16 + l16;
    for (int k0 = TKB; k0 <= K; k0 += TKB) {
        const bool last = (k0 == K);
#if USE_ASYNC_LDS
        if (!last) async_tile(k0, cur ^ 1);   // DMA next tile during compute
#else
        if (!last) fetch(k0);                 // overlap global loads with compute
#endif

        #pragma unroll
        for (int kk = 0; kk < TKB; kk += 32) {
            // A fragment: 16x32, K = g*16 + hi*8 + j
            v16bf afrag;
            {
                v8bf lo = *(const v8bf*)&As[cur][ra * PITCH + kk + hi * 8];
                v8bf h8 = *(const v8bf*)&As[cur][ra * PITCH + kk + 16 + hi * 8];
                #pragma unroll
                for (int j = 0; j < 8; ++j) { afrag[j] = lo[j]; afrag[j + 8] = h8[j]; }
            }
            // B fragments: col = l16, K = g*16 + hi*8 + j (stage all 4 first)
            v16bf bfr[4];
            #pragma unroll
            for (int f = 0; f < 4; ++f) {
                int rb = wcol * 64 + f * 16 + l16;
                v8bf b0 = *(const v8bf*)&Bs[cur][rb * PITCH + kk + hi * 8];
                v8bf b1 = *(const v8bf*)&Bs[cur][rb * PITCH + kk + 16 + hi * 8];
                #pragma unroll
                for (int j = 0; j < 8; ++j) { bfr[f][j] = b0[j]; bfr[f][j + 8] = b1[j]; }
            }
            // back-to-back WMMAs on disjoint accumulators
            #pragma unroll
            for (int f = 0; f < 4; ++f)
                acc[f] = __builtin_amdgcn_wmma_f32_16x16x32_bf16(
                    false, afrag, false, bfr[f], (short)0, acc[f], false, false);
        }

        if (!last) {
#if USE_ASYNC_LDS
            wait_async0();      // my DMA into buf^1 done
#else
            stash(cur ^ 1);     // fill other buffer while this one was consumed
#endif
            __syncthreads();
            cur ^= 1;
        }
    }

    // epilogue: C fragment element r -> row = 8*hi + r, col = l16
    #pragma unroll
    for (int f = 0; f < 4; ++f) {
        int ncol = tileN + wcol * 64 + f * 16 + l16;
        if constexpr (VST) {
            // V projection: store V^T  [b=m/1024][d=ncol][n=m%1024], 8 contiguous m per lane
            int m0 = tileM + wrow * 16 + hi * 8;
            int bb = m0 >> 10;
            int nt = m0 & 1023;
            float bvv = bias ? bias[ncol] : 0.0f;
            v8bf outv;
            #pragma unroll
            for (int r = 0; r < 8; ++r) outv[r] = (__bf16)(acc[f][r] * scale + bvv);
            *(v8bf*)((__bf16*)Cout + ((long)bb * 768 + ncol) * 1024 + nt) = outv;
        } else {
            #pragma unroll
            for (int r = 0; r < 8; ++r) {
                int mrow = tileM + wrow * 16 + hi * 8 + r;
                float x = acc[f][r] * scale;
                if (bias)     x += bias[ncol];
                if (rowbias)  x += rowbias[(long)(mrow / rowdiv) * N + ncol];
                if (residual) x += residual[(long)z * sC + (long)mrow * N + ncol];
                if constexpr (DOGELU)
                    x = 0.5f * x * (1.0f + erff(x * 0.70710678118654752f));
                long idx;
                if constexpr (PERM) {
                    // q_slot rows (i,b,s) -> [b][i*4+s][ncol]
                    int ii = mrow >> 7;
                    int bb = (mrow >> 2) & 31;
                    int ss = mrow & 3;
                    idx = (((long)bb * 64 + ii) * 4 + ss) * (long)N + ncol;
                } else {
                    idx = (long)z * sC + (long)mrow * N + ncol;
                }
                if constexpr (OUT_BF16) ((__bf16*)Cout)[idx] = (__bf16)x;
                else                    ((float*)Cout)[idx] = x;
            }
        }
    }
}

// ---------------------------------------------------------------------------
__global__ __launch_bounds__(256) void f2bf(const float* __restrict__ in,
                                            __bf16* __restrict__ out, long n) {
    long i = (long)blockIdx.x * 256 + threadIdx.x;
    if (i < n) out[i] = (__bf16)in[i];
}

__global__ __launch_bounds__(256) void slots_init(const float* __restrict__ noise,
        const float* __restrict__ mu, const float* __restrict__ sigma,
        float* __restrict__ slots) {
    long idx = (long)blockIdx.x * 256 + threadIdx.x;
    if (idx >= 6291456L) return;                 // I*B*S*D
    int sd = (int)(idx % 3072);                  // s*768 + d
    slots[idx] = mu[sd] + noise[idx] * sigma[sd];
}

__global__ __launch_bounds__(256) void qintent_kernel(const float* __restrict__ iq,
        const float* __restrict__ Wq_int, const float* __restrict__ bq_int,
        const float* __restrict__ bq_slot, float* __restrict__ qi_bias) {
    int idx = blockIdx.x * 256 + threadIdx.x;
    if (idx >= 64 * 768) return;
    int i = idx / 768, e = idx % 768;
    const float* q = iq + (long)i * 768;
    const float* w = Wq_int + (long)e * 768;
    float acc = 0.0f;
    for (int d = 0; d < 768; ++d) acc += q[d] * w[d];
    qi_bias[idx] = acc + bq_int[e] + bq_slot[e];
}

// wave-per-row softmax over 1024 cols, f32 in -> bf16 out
__global__ __launch_bounds__(256) void softmax_rows(const float* __restrict__ in,
                                                    __bf16* __restrict__ out) {
    int row  = blockIdx.x * 8 + (threadIdx.x >> 5);
    int lane = threadIdx.x & 31;
    const float* p = in + (long)row * 1024;
    float v[32];
    float mx = -3.0e38f;
    #pragma unroll
    for (int j = 0; j < 32; ++j) { v[j] = p[lane + 32 * j]; mx = fmaxf(mx, v[j]); }
    #pragma unroll
    for (int o = 16; o; o >>= 1) mx = fmaxf(mx, __shfl_xor(mx, o, 32));
    float sum = 0.0f;
    #pragma unroll
    for (int j = 0; j < 32; ++j) { v[j] = expf(v[j] - mx); sum += v[j]; }
    #pragma unroll
    for (int o = 16; o; o >>= 1) sum += __shfl_xor(sum, o, 32);
    float inv = 1.0f / sum;
    __bf16* q = out + (long)row * 1024;
    #pragma unroll
    for (int j = 0; j < 32; ++j) q[lane + 32 * j] = (__bf16)(v[j] * inv);
}

// slots = LN(slots + updates); h = LN2(slots)  (wave per row of 768)
__global__ __launch_bounds__(256) void add_ln_kernel(float* __restrict__ slots,
        const float* __restrict__ updP,
        const float* __restrict__ g1, const float* __restrict__ bb1,
        const float* __restrict__ g2, const float* __restrict__ bb2,
        __bf16* __restrict__ h) {
    int row  = blockIdx.x * 8 + (threadIdx.x >> 5);  // (i,b,s) flattened
    int lane = threadIdx.x & 31;
    int i = row >> 7, b = (row >> 2) & 31, s = row & 3;
    const float* up = updP + ((((long)b * 64 + i) * 4) + s) * 768;
    float* sp = slots + (long)row * 768;
    float x[24]; float sum = 0.0f, sq = 0.0f;
    #pragma unroll
    for (int j = 0; j < 24; ++j) { int d = lane + 32 * j; float v = sp[d] + up[d]; x[j] = v; sum += v; }
    #pragma unroll
    for (int o = 16; o; o >>= 1) sum += __shfl_xor(sum, o, 32);
    float mu = sum * (1.0f / 768.0f);
    #pragma unroll
    for (int j = 0; j < 24; ++j) { float c = x[j] - mu; sq += c * c; }
    #pragma unroll
    for (int o = 16; o; o >>= 1) sq += __shfl_xor(sq, o, 32);
    float rs = rsqrtf(sq * (1.0f / 768.0f) + 1e-5f);
    float y[24]; float sum2 = 0.0f, sq2 = 0.0f;
    #pragma unroll
    for (int j = 0; j < 24; ++j) {
        int d = lane + 32 * j;
        float v = (x[j] - mu) * rs * g1[d] + bb1[d];
        y[j] = v; sp[d] = v; sum2 += v;
    }
    #pragma unroll
    for (int o = 16; o; o >>= 1) sum2 += __shfl_xor(sum2, o, 32);
    float mu2 = sum2 * (1.0f / 768.0f);
    #pragma unroll
    for (int j = 0; j < 24; ++j) { float c = y[j] - mu2; sq2 += c * c; }
    #pragma unroll
    for (int o = 16; o; o >>= 1) sq2 += __shfl_xor(sq2, o, 32);
    float rs2 = rsqrtf(sq2 * (1.0f / 768.0f) + 1e-5f);
    __bf16* hp = h + (long)row * 768;
    #pragma unroll
    for (int j = 0; j < 24; ++j) {
        int d = lane + 32 * j;
        hp[d] = (__bf16)((y[j] - mu2) * rs2 * g2[d] + bb2[d]);
    }
}

// out[b,i] = sum_s <slots_norm[i,b,s], q_norm[i]>   (one block per (i,b), wave per s)
__global__ __launch_bounds__(128) void score_kernel(const float* __restrict__ slots,
        const float* __restrict__ iq, float* __restrict__ out) {
    int ib = blockIdx.x;
    int i = ib >> 5, b = ib & 31;
    int s = threadIdx.x >> 5;
    int lane = threadIdx.x & 31;
    const float* sp = slots + ((((long)i * 32 + b) * 4) + s) * 768;
    const float* qp = iq + (long)i * 768;
    float dot = 0.0f, ns = 0.0f, nq = 0.0f;
    #pragma unroll
    for (int j = 0; j < 24; ++j) {
        int d = lane + 32 * j;
        float sv = sp[d], qv = qp[d];
        dot += sv * qv; ns += sv * sv; nq += qv * qv;
    }
    #pragma unroll
    for (int o = 16; o; o >>= 1) {
        dot += __shfl_xor(dot, o, 32);
        ns  += __shfl_xor(ns,  o, 32);
        nq  += __shfl_xor(nq,  o, 32);
    }
    __shared__ float part[4];
    if (lane == 0)
        part[s] = dot / (fmaxf(sqrtf(ns), 1e-12f) * fmaxf(sqrtf(nq), 1e-12f));
    __syncthreads();
    if (threadIdx.x == 0)
        out[(long)b * 64 + i] = part[0] + part[1] + part[2] + part[3];
}

// ---------------------------------------------------------------------------
extern "C" void kernel_launch(void* const* d_in, const int* in_sizes, int n_in,
                              void* d_out, int out_size, void* d_ws, size_t ws_size,
                              hipStream_t stream) {
    const float* tokens  = (const float*)d_in[0];
    const float* iq      = (const float*)d_in[1];
    const float* noise   = (const float*)d_in[2];
    const float* slot_mu = (const float*)d_in[3];
    const float* slot_sg = (const float*)d_in[4];
    const float* Wq_slot = (const float*)d_in[5];
    const float* bq_slot = (const float*)d_in[6];
    const float* Wq_int  = (const float*)d_in[7];
    const float* bq_int  = (const float*)d_in[8];
    const float* Wk      = (const float*)d_in[9];
    const float* bk      = (const float*)d_in[10];
    const float* Wv      = (const float*)d_in[11];
    const float* bv      = (const float*)d_in[12];
    const float* lsg     = (const float*)d_in[13];
    const float* lsb     = (const float*)d_in[14];
    const float* lmg     = (const float*)d_in[15];
    const float* lmb     = (const float*)d_in[16];
    const float* W1      = (const float*)d_in[17];
    const float* b1      = (const float*)d_in[18];
    const float* W2      = (const float*)d_in[19];
    const float* b2      = (const float*)d_in[20];
    float* out = (float*)d_out;

    char* ws = (char*)d_ws;
    size_t off = 0;
    auto take = [&](size_t bytes) -> char* {
        char* p = ws + off;
        off = (off + bytes + 255) & ~(size_t)255;
        return p;
    };
    __bf16* tokbf  = (__bf16*)take(32768L * 768 * 2);
    __bf16* WkB    = (__bf16*)take(768L * 768 * 2);
    __bf16* WvB    = (__bf16*)take(768L * 768 * 2);
    __bf16* WqsB   = (__bf16*)take(768L * 768 * 2);
    __bf16* W1B    = (__bf16*)take(1536L * 768 * 2);
    __bf16* W2B    = (__bf16*)take(768L * 1536 * 2);
    __bf16* Kbf    = (__bf16*)take(32L * 1024 * 768 * 2);   // [b][n][d]
    __bf16* Vt     = (__bf16*)take(32L * 768 * 1024 * 2);   // [b][d][n]
    float*  qib    = (float*)take(64L * 768 * 4);           // bq_slot + q_intent
    float*  slots  = (float*)take(8192L * 768 * 4);         // [i][b][s][d]
    __bf16* slotsB = (__bf16*)take(8192L * 768 * 2);
    __bf16* qslotP = (__bf16*)take(8192L * 768 * 2);        // [b][i*4+s][d]
    float*  logits = (float*)take(32L * 256 * 1024 * 4);    // [b][i*4+s][n]
    __bf16* attn   = (__bf16*)take(32L * 256 * 1024 * 2);
    float*  updP   = (float*)take(8192L * 768 * 4);         // [b][i*4+s][d]
    __bf16* h      = (__bf16*)take(8192L * 768 * 2);        // [i][b][s][d]
    __bf16* h2     = (__bf16*)take(8192L * 1536 * 2);

    const float scl = 1.0f / sqrtf(768.0f);

    // precompute: bf16 conversions, q_intent bias, slot init, K/V projections
    f2bf<<<(25165824L + 255) / 256, 256, 0, stream>>>(tokens, tokbf, 25165824L);
    f2bf<<<(589824 + 255) / 256, 256, 0, stream>>>(Wk, WkB, 589824L);
    f2bf<<<(589824 + 255) / 256, 256, 0, stream>>>(Wv, WvB, 589824L);
    f2bf<<<(589824 + 255) / 256, 256, 0, stream>>>(Wq_slot, WqsB, 589824L);
    f2bf<<<(1179648 + 255) / 256, 256, 0, stream>>>(W1, W1B, 1179648L);
    f2bf<<<(1179648 + 255) / 256, 256, 0, stream>>>(W2, W2B, 1179648L);
    qintent_kernel<<<192, 256, 0, stream>>>(iq, Wq_int, bq_int, bq_slot, qib);
    slots_init<<<(6291456L + 255) / 256, 256, 0, stream>>>(noise, slot_mu, slot_sg, slots);

    // K = tokens*Wk^T + bk  -> bf16 [b][n][d]
    wmma_gemm<true, false, false, false><<<dim3(512, 6, 1), 256, 0, stream>>>(
        tokbf, WkB, Kbf, 32768, 768, 768, 0, 0, 0, bk, nullptr, 1, nullptr, 1.0f);
    // V = tokens*Wv^T + bv  -> bf16 transposed [b][d][n]
    wmma_gemm<true, false, false, true><<<dim3(512, 6, 1), 256, 0, stream>>>(
        tokbf, WvB, Vt, 32768, 768, 768, 0, 0, 0, bv, nullptr, 1, nullptr, 1.0f);

    for (int it = 0; it < 3; ++it) {
        f2bf<<<(6291456L + 255) / 256, 256, 0, stream>>>(slots, slotsB, 6291456L);
        // q_slot = slots*Wq_slot^T + (bq_slot + q_intent[i])  -> permuted bf16 [b][i*4+s][d]
        wmma_gemm<true, false, true, false><<<dim3(128, 6, 1), 256, 0, stream>>>(
            slotsB, WqsB, qslotP, 8192, 768, 768, 0, 0, 0, nullptr, qib, 128, nullptr, 1.0f);
        // logits[b] = qslotP[b] * K[b]^T * scale   (batched over b)
        wmma_gemm<false, false, false, false><<<dim3(4, 8, 32), 256, 0, stream>>>(
            qslotP, Kbf, logits, 256, 1024, 768,
            256L * 768, 1024L * 768, 256L * 1024, nullptr, nullptr, 1, nullptr, scl);
        softmax_rows<<<1024, 256, 0, stream>>>(logits, attn);
        // updates[b] = attn[b] * V[b]   (Bt = V^T rows)
        wmma_gemm<false, false, false, false><<<dim3(4, 6, 32), 256, 0, stream>>>(
            attn, Vt, updP, 256, 768, 1024,
            256L * 1024, 768L * 1024, 256L * 768, nullptr, nullptr, 1, nullptr, 1.0f);
        add_ln_kernel<<<1024, 256, 0, stream>>>(slots, updP, lsg, lsb, lmg, lmb, h);
        // h2 = gelu(h*W1^T + b1)
        wmma_gemm<true, true, false, false><<<dim3(128, 12, 1), 256, 0, stream>>>(
            h, W1B, h2, 8192, 1536, 768, 0, 0, 0, b1, nullptr, 1, nullptr, 1.0f);
        // slots = slots + h2*W2^T + b2
        wmma_gemm<false, false, false, false><<<dim3(128, 6, 1), 256, 0, stream>>>(
            h2, W2B, slots, 8192, 768, 1536, 0, 0, 0, b2, nullptr, 1, slots, 1.0f);
    }
    score_kernel<<<2048, 128, 0, stream>>>(slots, iq, out);
}
